// MLPEncoder_13735305413086
// MI455X (gfx1250) — matrix operand order repaired and checked by reading, not compiled, not run on hardware
//
#include <hip/hip_runtime.h>
#include <hip/hip_bf16.h>

// ---------------------------------------------------------------------------
// NRI MLP encoder for MI455X (gfx1250, wave32, WMMA).
// Strategy: all GEMMs via v_wmma_f32_16x16x32_f16 (f16 in, f32 accum).
// node2edge gathers fused into GEMM A-loads (node features are L2-resident);
// mlp2 BN folded to per-feature affine applied at consumers; mlp4 BN folded
// into the final 256->4 FC. Intermediate edge tensors stored f16.
// Workspace requirement: ~535 MB.
// ---------------------------------------------------------------------------

typedef __attribute__((ext_vector_type(16))) _Float16 half16;
typedef __attribute__((ext_vector_type(8)))  _Float16 half8;
typedef __attribute__((ext_vector_type(8)))  float    float8;

constexpr int   Bc   = 32;
constexpr int   Nc   = 128;
constexpr int   NINc = 256;
constexpr int   Hc   = 256;           // all hidden/out widths are 256
constexpr int   Ec   = Nc * (Nc - 1); // 16256
constexpr int   M1   = Bc * Nc;       // 4096 node rows
constexpr long long M2 = (long long)Bc * Ec; // 520192 edge rows
constexpr float BN_EPS = 1e-5f;

// ------------------------------ small utility kernels ----------------------

__global__ void zero_f32_kernel(float* p, int n) {
    int i = blockIdx.x * blockDim.x + threadIdx.x;
    if (i < n) p[i] = 0.f;
}

__global__ void f32_to_f16_kernel(const float* __restrict__ x,
                                  _Float16* __restrict__ y, int n) {
    int i = blockIdx.x * blockDim.x + threadIdx.x;
    if (i < n) y[i] = (_Float16)x[i];
}

// W [K,256] f32 row-major  ->  Wt [256,K] f16 (transposed for B-fragments)
__global__ void wt_cvt_kernel(const float* __restrict__ W,
                              _Float16* __restrict__ Wt, int K) {
    int idx = blockIdx.x * blockDim.x + threadIdx.x; // coalesced over n
    if (idx >= K * 256) return;
    int k = idx >> 8, n = idx & 255;
    Wt[(size_t)n * K + k] = (_Float16)W[idx];
}

// Recover sender/receiver node index from one-hot rel matrices.
__global__ void edge_idx_kernel(const float* __restrict__ rel_send,
                                const float* __restrict__ rel_rec,
                                int* __restrict__ sIdx, int* __restrict__ rIdx) {
    int e = blockIdx.x * blockDim.x + threadIdx.x;
    if (e >= Ec) return;
    int s = 0, r = 0;
    for (int n = 0; n < Nc; ++n) {
        if (rel_send[(size_t)e * Nc + n] > 0.5f) s = n;
        if (rel_rec [(size_t)e * Nc + n] > 0.5f) r = n;
    }
    sIdx[e] = s; rIdx[e] = r;
}

// ------------------------------ WMMA GEMM ----------------------------------
// C[M,256] = ELU(A[M,K] @ W[K,256] + bias), f16 in/out, f32 accumulate.
// MODE 0: A is a plain f16 [M,K] row-major tensor.
// MODE 2: K=512 node2edge gather: k<256 -> Xn[b, send(e)], else Xn[b, recv(e)].
// MODE 3: K=768: [Xn3 send | Xn3 recv | affine(Xe[row]) ] (mlp2-BN folded).
// One wave computes one 16x16 tile; 8 waves/block share the same 16 A-rows.
template <int MODE>
__global__ void gemm_wmma_f16(const _Float16* __restrict__ A,
                              const _Float16* __restrict__ Xn,
                              const _Float16* __restrict__ Xe,
                              const float* __restrict__ sAff,
                              const float* __restrict__ tAff,
                              const int* __restrict__ sendIdx,
                              const int* __restrict__ recvIdx,
                              const _Float16* __restrict__ Wt,   // [256,K]
                              const float* __restrict__ bias,    // [256]
                              _Float16* __restrict__ Cout,       // [M,256]
                              int K) {
    const int lane = threadIdx.x & 31;
    const int wave = threadIdx.x >> 5;
    const int m0 = blockIdx.x * 16;
    const int n0 = blockIdx.y * 128 + wave * 16;
    const int rA = m0 + (lane & 15);          // A-row this lane supplies

    int bIdx = 0, sNode = 0, rNode = 0;
    if constexpr (MODE != 0) {
        bIdx = rA / Ec;
        const int eIdx = rA - bIdx * Ec;
        sNode = sendIdx[eIdx];
        rNode = recvIdx[eIdx];
    }

    // per-lane A K-offsets: +{0..15} contiguous, base 16 for upper half-wave
    const int kbA = (lane >= 16) ? 16 : 0;
    const int kbB = (lane >= 16) ? 16 : 0;
    const _Float16* wrow = Wt + (size_t)(n0 + (lane & 15)) * K;

    auto loadA8 = [&](int k) -> half8 {
        if constexpr (MODE == 0) {
            return *(const half8*)(A + (size_t)rA * K + k);
        } else if constexpr (MODE == 2) {
            const int node = (k < 256) ? sNode : rNode;
            return *(const half8*)(Xn + ((size_t)bIdx * Nc + node) * 256 + (k & 255));
        } else { // MODE 3
            if (k < 512) {
                const int node = (k < 256) ? sNode : rNode;
                return *(const half8*)(Xn + ((size_t)bIdx * Nc + node) * 256 + (k & 255));
            } else {
                const int kk = k - 512;
                half8 h = *(const half8*)(Xe + (size_t)rA * 256 + kk);
                half8 o;
#pragma unroll
                for (int j = 0; j < 8; ++j)
                    o[j] = (_Float16)(sAff[kk + j] * (float)h[j] + tAff[kk + j]);
                return o;
            }
        }
    };

    float8 c = {};
    for (int k0 = 0; k0 < K; k0 += 32) {
        half8 g0 = loadA8(k0 + kbA);
        half8 g1 = loadA8(k0 + kbA + 8);
        half16 a;
#pragma unroll
        for (int j = 0; j < 8; ++j) { a[j] = g0[j]; a[8 + j] = g1[j]; }
        half16 b = *(const half16*)(wrow + k0 + kbB);
        c = __builtin_amdgcn_wmma_f32_16x16x32_f16(
                false, a, false, b, (short)0, c, false, false);
    }

    // epilogue: bias + ELU, store f16
    const int col = n0 + (lane & 15);
    const float bs = bias[col];
    const int rbase = m0 + ((lane >= 16) ? 8 : 0);
#pragma unroll
    for (int r = 0; r < 8; ++r) {
        float v = c[r] + bs;
        v = (v > 0.f) ? v : (__expf(v) - 1.f);
        Cout[(size_t)(rbase + r) * 256 + col] = (_Float16)v;
    }
}

// ------------------------------ BatchNorm pieces ---------------------------

// column sums / sumsq over [M,256]; 256 rows per block, atomics per block.
__global__ void colstats_kernel(const _Float16* __restrict__ X,
                                float* __restrict__ st /*[512]*/) {
    const int f = threadIdx.x;
    const size_t r0 = (size_t)blockIdx.x * 256;
    float s = 0.f, q = 0.f;
    for (int r = 0; r < 256; ++r) {
        float v = (float)X[(r0 + r) * 256 + f];
        s += v; q += v * v;
    }
    atomicAdd(&st[f], s);
    atomicAdd(&st[256 + f], q);
}

// BN(h) = h*s + t per feature
__global__ void bn_coeffs_kernel(const float* __restrict__ st,
                                 const float* __restrict__ gamma,
                                 const float* __restrict__ beta,
                                 float* __restrict__ s, float* __restrict__ t,
                                 float invM) {
    const int f = threadIdx.x;
    const float mean = st[f] * invM;
    const float var  = st[256 + f] * invM - mean * mean;
    const float sc   = gamma[f] * rsqrtf(var + BN_EPS);
    s[f] = sc;
    t[f] = beta[f] - mean * sc;
}

__global__ void bn_apply_kernel(const _Float16* __restrict__ X,
                                const float* __restrict__ s,
                                const float* __restrict__ t,
                                _Float16* __restrict__ Y, int total) {
    int i = blockIdx.x * blockDim.x + threadIdx.x;
    if (i >= total) return;
    int f = i & 255;
    Y[i] = (_Float16)((float)X[i] * s[f] + t[f]);
}

// ------------------------------ graph ops ----------------------------------

// edge2node: xn[b,n,f] = (1/N) * sum_{i != n} BNaffine(g2[b, e(i->n), f])
__global__ void edge2node_kernel(const _Float16* __restrict__ g2,
                                 const float* __restrict__ s,
                                 const float* __restrict__ t,
                                 _Float16* __restrict__ xn) {
    const int f = threadIdx.x;          // 256 features
    const int bn = blockIdx.x;          // B*N blocks
    const int b = bn >> 7, n = bn & 127;
    float acc = 0.f;
    for (int i = 0; i < Nc; ++i) {
        if (i == n) continue;
        const int e = i * (Nc - 1) + ((n < i) ? n : n - 1);
        acc += (float)g2[((size_t)b * Ec + e) * 256 + f];
    }
    const float val = (s[f] * acc + (float)(Nc - 1) * t[f]) * (1.f / (float)Nc);
    xn[((size_t)b * Nc + n) * 256 + f] = (_Float16)val;
}

// fold mlp4-BN + fc into W'[4,256], b'[4]  (single block of 256)
__global__ void fold_fc_kernel(const float* __restrict__ st,
                               const float* __restrict__ gamma,
                               const float* __restrict__ beta,
                               const float* __restrict__ fcW,  // [256,4]
                               const float* __restrict__ fcb,  // [4]
                               float* __restrict__ Wp,         // [4,256]
                               float* __restrict__ bp,         // [4]
                               float invM) {
    const int k = threadIdx.x;
    if (k < 4) bp[k] = fcb[k];
    __syncthreads();
    const float mean = st[k] * invM;
    const float var  = st[256 + k] * invM - mean * mean;
    const float s    = gamma[k] * rsqrtf(var + BN_EPS);
    const float t    = beta[k] - mean * s;
#pragma unroll
    for (int c = 0; c < 4; ++c) {
        Wp[c * 256 + k] = s * fcW[k * 4 + c];
        atomicAdd(&bp[c], t * fcW[k * 4 + c]);
    }
}

// out[m, 0..3] = h4[m,:] @ W'^T + b'
__global__ void final_fc_kernel(const _Float16* __restrict__ H,
                                const float* __restrict__ Wp,
                                const float* __restrict__ bp,
                                float* __restrict__ out) {
    const size_t m = (size_t)blockIdx.x * blockDim.x + threadIdx.x;
    if (m >= (size_t)M2) return;
    float a0 = bp[0], a1 = bp[1], a2 = bp[2], a3 = bp[3];
    const _Float16* h = H + m * 256;
    for (int k = 0; k < 256; k += 8) {
        half8 v = *(const half8*)(h + k);
#pragma unroll
        for (int j = 0; j < 8; ++j) {
            const float x = (float)v[j];
            a0 += x * Wp[k + j];
            a1 += x * Wp[256 + k + j];
            a2 += x * Wp[512 + k + j];
            a3 += x * Wp[768 + k + j];
        }
    }
    float4 o; o.x = a0; o.y = a1; o.z = a2; o.w = a3;
    *(float4*)(out + m * 4) = o;
}

// ------------------------------ driver -------------------------------------

extern "C" void kernel_launch(void* const* d_in, const int* in_sizes, int n_in,
                              void* d_out, int out_size, void* d_ws, size_t ws_size,
                              hipStream_t stream) {
    (void)in_sizes; (void)n_in; (void)out_size;

    const float* inp      = (const float*)d_in[0];
    const float* rel_rec  = (const float*)d_in[1];
    const float* rel_send = (const float*)d_in[2];
    // per-MLP params: W1,b1,W2,b2,gamma,beta at base 3 + 6*layer
    const float* W1[4], *b1[4], *W2[4], *b2[4], *ga[4], *be[4];
    for (int l = 0; l < 4; ++l) {
        W1[l] = (const float*)d_in[3 + 6 * l + 0];
        b1[l] = (const float*)d_in[3 + 6 * l + 1];
        W2[l] = (const float*)d_in[3 + 6 * l + 2];
        b2[l] = (const float*)d_in[3 + 6 * l + 3];
        ga[l] = (const float*)d_in[3 + 6 * l + 4];
        be[l] = (const float*)d_in[3 + 6 * l + 5];
    }
    const float* fcW = (const float*)d_in[27];
    const float* fcb = (const float*)d_in[28];
    float* out = (float*)d_out;

    // ---- workspace bump allocator (needs ~535 MB) ----
    unsigned char* base = (unsigned char*)d_ws;
    size_t off = 0;
    auto alloc = [&](size_t bytes) -> unsigned char* {
        unsigned char* p = base + off;
        off = (off + bytes + 255) & ~(size_t)255;
        return p;
    };
    const size_t edgeBufBytes = (size_t)M2 * 256 * sizeof(_Float16); // 266 MB
    _Float16* bufA   = (_Float16*)alloc(edgeBufBytes);   // edge ping
    _Float16* bufB   = (_Float16*)alloc(edgeBufBytes);   // edge pong
    const int Kin[4]  = {NINc, 2 * Hc, Hc, 3 * Hc};
    _Float16* Wt1[4], *Wt2[4];
    for (int l = 0; l < 4; ++l) {
        Wt1[l] = (_Float16*)alloc((size_t)Kin[l] * 256 * sizeof(_Float16));
        Wt2[l] = (_Float16*)alloc((size_t)Hc * 256 * sizeof(_Float16));
    }
    _Float16* A0     = (_Float16*)alloc((size_t)M1 * 256 * sizeof(_Float16));
    _Float16* nodeT1 = (_Float16*)alloc((size_t)M1 * 256 * sizeof(_Float16));
    _Float16* nodeT2 = (_Float16*)alloc((size_t)M1 * 256 * sizeof(_Float16));
    _Float16* xnode1 = (_Float16*)alloc((size_t)M1 * 256 * sizeof(_Float16));
    _Float16* xnE2N  = (_Float16*)alloc((size_t)M1 * 256 * sizeof(_Float16));
    _Float16* xnode3 = (_Float16*)alloc((size_t)M1 * 256 * sizeof(_Float16));
    int*   sendIdx = (int*)alloc(Ec * sizeof(int));
    int*   recvIdx = (int*)alloc(Ec * sizeof(int));
    float* stats   = (float*)alloc(4 * 512 * sizeof(float)); // st[l] = stats+512*l
    float* sAff    = (float*)alloc(3 * 256 * sizeof(float)); // s for mlp1..3
    float* tAff    = (float*)alloc(3 * 256 * sizeof(float));
    float* Wp      = (float*)alloc(1024 * sizeof(float));
    float* bp      = (float*)alloc(16 * sizeof(float));
    (void)ws_size;

    // ---- prep: zero stats, convert weights/inputs, edge indices ----
    zero_f32_kernel<<<dim3(8), dim3(256), 0, stream>>>(stats, 4 * 512);
    for (int l = 0; l < 4; ++l) {
        wt_cvt_kernel<<<dim3(Kin[l]), dim3(256), 0, stream>>>(W1[l], Wt1[l], Kin[l]);
        wt_cvt_kernel<<<dim3(Hc),     dim3(256), 0, stream>>>(W2[l], Wt2[l], Hc);
    }
    f32_to_f16_kernel<<<dim3(M1 * 256 / 256), dim3(256), 0, stream>>>(inp, A0, M1 * 256);
    edge_idx_kernel<<<dim3((Ec + 255) / 256), dim3(256), 0, stream>>>(rel_send, rel_rec,
                                                                      sendIdx, recvIdx);

    const dim3 blk(256);
    const dim3 gN(M1 / 16, 2);                 // node-row GEMM grid
    const dim3 gE((unsigned)(M2 / 16), 2);     // edge-row GEMM grid

    // ---- mlp1 (node rows) ----
    gemm_wmma_f16<0><<<gN, blk, 0, stream>>>(A0, nullptr, nullptr, nullptr, nullptr,
                                             nullptr, nullptr, Wt1[0], b1[0], nodeT1, NINc);
    gemm_wmma_f16<0><<<gN, blk, 0, stream>>>(nodeT1, nullptr, nullptr, nullptr, nullptr,
                                             nullptr, nullptr, Wt2[0], b2[0], nodeT2, Hc);
    colstats_kernel<<<dim3(M1 / 256), blk, 0, stream>>>(nodeT2, stats + 0 * 512);
    bn_coeffs_kernel<<<dim3(1), blk, 0, stream>>>(stats + 0 * 512, ga[0], be[0],
                                                  sAff + 0, tAff + 0, 1.f / (float)M1);
    bn_apply_kernel<<<dim3(M1), blk, 0, stream>>>(nodeT2, sAff + 0, tAff + 0,
                                                  xnode1, M1 * 256);

    // ---- mlp2 (edge rows, node2edge gather fused) ----
    gemm_wmma_f16<2><<<gE, blk, 0, stream>>>(nullptr, xnode1, nullptr, nullptr, nullptr,
                                             sendIdx, recvIdx, Wt1[1], b1[1], bufA, 2 * Hc);
    gemm_wmma_f16<0><<<gE, blk, 0, stream>>>(bufA, nullptr, nullptr, nullptr, nullptr,
                                             nullptr, nullptr, Wt2[1], b2[1], bufB, Hc);
    colstats_kernel<<<dim3((unsigned)(M2 / 256)), blk, 0, stream>>>(bufB, stats + 1 * 512);
    bn_coeffs_kernel<<<dim3(1), blk, 0, stream>>>(stats + 1 * 512, ga[1], be[1],
                                                  sAff + 256, tAff + 256, 1.f / (float)M2);
    // x_skip = bufB * s2 + t2 (applied on the fly by its consumers)

    // ---- edge2node + mlp3 (node rows) ----
    edge2node_kernel<<<dim3(Bc * Nc), blk, 0, stream>>>(bufB, sAff + 256, tAff + 256, xnE2N);
    gemm_wmma_f16<0><<<gN, blk, 0, stream>>>(xnE2N, nullptr, nullptr, nullptr, nullptr,
                                             nullptr, nullptr, Wt1[2], b1[2], nodeT1, Hc);
    gemm_wmma_f16<0><<<gN, blk, 0, stream>>>(nodeT1, nullptr, nullptr, nullptr, nullptr,
                                             nullptr, nullptr, Wt2[2], b2[2], nodeT2, Hc);
    colstats_kernel<<<dim3(M1 / 256), blk, 0, stream>>>(nodeT2, stats + 2 * 512);
    bn_coeffs_kernel<<<dim3(1), blk, 0, stream>>>(stats + 2 * 512, ga[2], be[2],
                                                  sAff + 512, tAff + 512, 1.f / (float)M1);
    bn_apply_kernel<<<dim3(M1), blk, 0, stream>>>(nodeT2, sAff + 512, tAff + 512,
                                                  xnode3, M1 * 256);

    // ---- mlp4 (edge rows; gather [send3|recv3|affine(x_skip)]) ----
    gemm_wmma_f16<3><<<gE, blk, 0, stream>>>(nullptr, xnode3, bufB, sAff + 256, tAff + 256,
                                             sendIdx, recvIdx, Wt1[3], b1[3], bufA, 3 * Hc);
    gemm_wmma_f16<0><<<gE, blk, 0, stream>>>(bufA, nullptr, nullptr, nullptr, nullptr,
                                             nullptr, nullptr, Wt2[3], b2[3], bufB, Hc);
    colstats_kernel<<<dim3((unsigned)(M2 / 256)), blk, 0, stream>>>(bufB, stats + 3 * 512);

    // ---- BN4 folded into fc, then final streaming FC ----
    fold_fc_kernel<<<dim3(1), blk, 0, stream>>>(stats + 3 * 512, ga[3], be[3],
                                                fcW, fcb, Wp, bp, 1.f / (float)M2);
    final_fc_kernel<<<dim3((unsigned)((M2 + 255) / 256)), blk, 0, stream>>>(bufB, Wp, bp, out);
}
